// WarpAwareLocalRefiner_27281632264670
// MI455X (gfx1250) — matrix operand order repaired and checked by reading, not compiled, not run on hardware
//
#include <hip/hip_runtime.h>
#include <hip/hip_bf16.h>
#include <math.h>

typedef __attribute__((ext_vector_type(16))) _Float16 v16h;
typedef __attribute__((ext_vector_type(8)))  _Float16 v8h;
typedef __attribute__((ext_vector_type(8)))  float    v8f;

#define BATCH 2
#define CH    256
#define HH    128
#define WW    128
#define HW    (HH*WW)            // 16384
#define NPIX  (BATCH*HW)         // 32768
#define K1P   800                // 780 padded to 32-multiple
#define K1R   780
#define K2P   576                // 64*9, already 32-multiple
#define HID   64

// ---------------- helpers ----------------
__device__ __forceinline__ float ntn0(float v) {            // nan/±inf -> 0
    return isfinite(v) ? v : 0.0f;
}
__device__ __forceinline__ float route_clip(float v) {      // nan->0, then clip(-1.5,1.5)
    if (isnan(v)) return 0.0f;
    return fminf(fmaxf(v, -1.5f), 1.5f);
}
__device__ __forceinline__ float conf_clip(float v) {       // nan->0, clip(0,1)
    if (isnan(v)) return 0.0f;
    return fminf(fmaxf(v, 0.0f), 1.0f);
}
__device__ __forceinline__ float gelu_exact(float x) {
    return 0.5f * x * (1.0f + erff(x * 0.70710678118654752f));
}

struct BL { int i00, i01, i10, i11; float wx, wy; };
__device__ __forceinline__ BL mkbl(float gx, float gy) {
    float fx = (gx + 1.0f) * 64.0f - 0.5f;   // W*0.5 = 64
    float fy = (gy + 1.0f) * 64.0f - 0.5f;
    fx = fminf(fmaxf(fx, 0.0f), 127.0f);
    fy = fminf(fmaxf(fy, 0.0f), 127.0f);
    float x0f = floorf(fx), y0f = floorf(fy);
    int x0 = (int)x0f, y0 = (int)y0f;
    int x1 = (x0 + 1 < 127) ? x0 + 1 : 127;
    int y1 = (y0 + 1 < 127) ? y0 + 1 : 127;
    BL r;
    r.wx = fx - x0f; r.wy = fy - y0f;
    r.i00 = y0 * WW + x0; r.i01 = y0 * WW + x1;
    r.i10 = y1 * WW + x0; r.i11 = y1 * WW + x1;
    return r;
}
__device__ __forceinline__ float blerp(float v00, float v01, float v10, float v11,
                                       float wx, float wy) {
    float top = v00 + wx * (v01 - v00);
    float bot = v10 + wx * (v11 - v10);
    return top + wy * (bot - top);
}

// ---------------- 1) weight repack fp32 -> f16 (im2col k-order) ----------------
__global__ void k_prep_weights(const float* __restrict__ w1, const float* __restrict__ w2,
                               const float* __restrict__ w3,
                               _Float16* __restrict__ W1, _Float16* __restrict__ W2,
                               _Float16* __restrict__ W3) {
    int i = blockIdx.x * blockDim.x + threadIdx.x;
    if (i < HID * K1P) {
        int m = i / K1P, k = i % K1P;
        W1[i] = (_Float16)(k < K1R ? w1[m * K1R + k] : 0.0f);
    }
    int j = i - HID * K1P;
    if (j >= 0 && j < HID * K2P) {
        int m = j / K2P, k = j % K2P;
        int t = k / HID, c = k % HID;                 // k = t*64 + c ; t = ky*3+kx
        W2[j] = (_Float16)w2[(m * HID + c) * 9 + t];
        W3[j] = (_Float16)w3[(m * HID + c) * 9 + t];
    }
}

// ---------------- 2) gather stage: build X1[pixel][800] (f16) ----------------
__global__ __launch_bounds__(256)
void k_features(const float* __restrict__ fA, const float* __restrict__ fB,
                const float* __restrict__ cw, const float* __restrict__ cf,
                _Float16* __restrict__ X) {
    int p = blockIdx.x * blockDim.x + threadIdx.x;
    if (p >= NPIX) return;
    int b = p >> 14, pix = p & (HW - 1);
    int y = pix >> 7, x = pix & (WW - 1);

    float rx = route_clip(cw[2 * p + 0]);
    float ry = route_clip(cw[2 * p + 1]);

    // grid 0 = route (raw feat_B); grids 1..9 = clip(route + off) (nan_to_num'd feat_B)
    BL bl[10];
    bl[0] = mkbl(rx, ry);
#pragma unroll
    for (int t = 0; t < 9; ++t) {
        float ox = (float)(t % 3 - 1) * (1.0f / 64.0f);   // 2*dx/W
        float oy = (float)(t / 3 - 1) * (1.0f / 64.0f);   // 2*dy/H
        float gx = fminf(fmaxf(rx + ox, -1.5f), 1.5f);
        float gy = fminf(fmaxf(ry + oy, -1.5f), 1.5f);
        bl[t + 1] = mkbl(gx, gy);
    }

    const float* fAb = fA + (size_t)b * CH * HW;
    const float* fBb = fB + (size_t)b * CH * HW;
    _Float16* Xrow = X + (size_t)p * K1P;

    float fa2 = 0.0f;
    float dot[9], s2[9];
#pragma unroll
    for (int t = 0; t < 9; ++t) { dot[t] = 0.0f; s2[t] = 0.0f; }

    for (int c = 0; c < CH; ++c) {
        const float* plane = fBb + (size_t)c * HW;
        float fa_raw = fAb[(size_t)c * HW + pix];

        // warped_B: raw feat_B at route
        float warp = blerp(plane[bl[0].i00], plane[bl[0].i01],
                           plane[bl[0].i10], plane[bl[0].i11], bl[0].wx, bl[0].wy);
        Xrow[c]          = (_Float16)fa_raw;
        Xrow[CH + c]     = (_Float16)warp;
        Xrow[2 * CH + c] = (_Float16)fabsf(fa_raw - warp);

        float fa = ntn0(fa_raw);
        fa2 += fa * fa;
#pragma unroll
        for (int t = 0; t < 9; ++t) {
            const BL& g = bl[t + 1];
            float s = blerp(ntn0(plane[g.i00]), ntn0(plane[g.i01]),
                            ntn0(plane[g.i10]), ntn0(plane[g.i11]), g.wx, g.wy);
            dot[t] += fa * s;
            s2[t]  += s * s;
        }
    }

    float sx = ((float)x + 0.5f) * (2.0f / WW) - 1.0f;
    float sy = ((float)y + 0.5f) * (2.0f / HH) - 1.0f;
    Xrow[768] = (_Float16)(rx - sx);
    Xrow[769] = (_Float16)(ry - sy);
    Xrow[770] = (_Float16)conf_clip(cf[p]);
    float fan = fmaxf(sqrtf(fa2), 1e-12f);
#pragma unroll
    for (int t = 0; t < 9; ++t) {
        float sn = fmaxf(sqrtf(s2[t]), 1e-12f);
        Xrow[771 + t] = (_Float16)(dot[t] / (fan * sn));
    }
#pragma unroll
    for (int k = K1R; k < K1P; ++k) Xrow[k] = (_Float16)0.0f;
}

// ---------------- 3) WMMA GEMM: Y[n][64] = X[n][Kp] . W[64][Kp]^T ----------------
// block = 256 thr = 8 waves; wave w: M-tile = (w&3)*16, N-tile = blk*32 + (w>>2)*16
__global__ __launch_bounds__(256)
void k_gemm(const _Float16* __restrict__ X, const _Float16* __restrict__ W,
            float* __restrict__ Y, int Kp) {
    int wave = threadIdx.x >> 5;
    int lane = threadIdx.x & 31;
    int mt = (wave & 3) * 16;
    int nb = blockIdx.x * 32 + (wave >> 2) * 16;
    int lm = lane & 15;
    int hi = lane >> 4;

    // A-fragment (16x32), contiguous k order used here (same as B): lanes<16 -> K 0..15, lanes>=16 -> K 16..31
    const _Float16* Arow = W + (size_t)(mt + lm) * Kp + hi * 16;
    // B-fragment (32x16): lane n = lm, 16 contiguous K per lane, lanes>=16 -> K+16
    const _Float16* Brow = X + (size_t)(nb + lm) * Kp + hi * 16;

    v8f acc = {};
    for (int kb = 0; kb < Kp; kb += 32) {
        v8h a0 = *(const v8h*)(Arow + kb);
        v8h a1 = *(const v8h*)(Arow + kb + 8);
        v16h a;
#pragma unroll
        for (int i = 0; i < 8; ++i) { a[i] = a0[i]; a[8 + i] = a1[i]; }
        v16h bf = *(const v16h*)(Brow + kb);
        acc = __builtin_amdgcn_wmma_f32_16x16x32_f16(
            false, a, false, bf, (short)0, acc, false, false);
    }
    // D 16x16 f32: VGPR r -> row mt + hi*8 + r, col nb + lm
    float* Yp = Y + (size_t)(nb + lm) * HID + mt + hi * 8;
#pragma unroll
    for (int r = 0; r < 8; ++r) Yp[r] = acc[r];
}

// ---------------- 4) GroupNorm stats: one block per (batch, group) ----------------
__global__ __launch_bounds__(256)
void k_stats(const float* __restrict__ Y, float* __restrict__ st) {
    int b = blockIdx.x >> 3, g = blockIdx.x & 7;
    __shared__ float ss[256], ss2[256];
    float s = 0.0f, s2 = 0.0f;
    const float* Yb = Y + (size_t)b * HW * HID + g * 8;
    for (int idx = threadIdx.x; idx < HW * 8; idx += 256) {
        int pix = idx >> 3, c = idx & 7;
        float v = Yb[(size_t)pix * HID + c];
        s += v; s2 += v * v;
    }
    ss[threadIdx.x] = s; ss2[threadIdx.x] = s2;
    __syncthreads();
    for (int o = 128; o > 0; o >>= 1) {
        if (threadIdx.x < o) {
            ss[threadIdx.x]  += ss[threadIdx.x + o];
            ss2[threadIdx.x] += ss2[threadIdx.x + o];
        }
        __syncthreads();
    }
    if (threadIdx.x == 0) {
        const float invn = 1.0f / (float)(HW * 8);
        float mean = ss[0] * invn;
        float var  = ss2[0] * invn - mean * mean;
        st[blockIdx.x * 2 + 0] = mean;
        st[blockIdx.x * 2 + 1] = rsqrtf(fmaxf(var, 0.0f) + 1e-5f);
    }
}

// ---------------- 5) GN + GELU + im2col -> X[pixel][576] f16 ----------------
__global__ __launch_bounds__(256)
void k_act_im2col(const float* __restrict__ Y, const float* __restrict__ st,
                  const float* __restrict__ gamma, const float* __restrict__ beta,
                  _Float16* __restrict__ X) {
    int p = blockIdx.x * 4 + (threadIdx.x >> 6);
    int c = threadIdx.x & 63;
    int b = p >> 14, pix = p & (HW - 1);
    int y = pix >> 7, x = pix & (WW - 1);
    int sidx = (b * 8 + (c >> 3)) * 2;
    float mean = st[sidx], inv = st[sidx + 1];
    float ga = gamma[c], be = beta[c];
    const float* Yb = Y + (size_t)b * HW * HID + c;
    _Float16* Xrow = X + (size_t)p * K2P + c;
#pragma unroll
    for (int t = 0; t < 9; ++t) {
        int yy = y + t / 3 - 1, xx = x + t % 3 - 1;
        float v = 0.0f;
        if (yy >= 0 && yy < HH && xx >= 0 && xx < WW) {
            float raw = Yb[(size_t)(yy * WW + xx) * HID];
            v = gelu_exact((raw - mean) * inv * ga + be);
        }
        Xrow[t * HID] = (_Float16)v;
    }
}

// ---------------- 6) epilogue: delta/conf 3x3 convs (M=3, scalar) + outputs ----------------
__global__ __launch_bounds__(256)
void k_final(const float* __restrict__ Y, const float* __restrict__ st,
             const float* __restrict__ g3, const float* __restrict__ b3,
             const float* __restrict__ cw, const float* __restrict__ cf,
             const float* __restrict__ dw, const float* __restrict__ db,
             const float* __restrict__ cfw, const float* __restrict__ cfb,
             float* __restrict__ out) {
    int p = blockIdx.x * blockDim.x + threadIdx.x;
    if (p >= NPIX) return;
    int b = p >> 14, pix = p & (HW - 1);
    int y = pix >> 7, x = pix & (WW - 1);

    float mg[8], ig[8];
#pragma unroll
    for (int g = 0; g < 8; ++g) {
        mg[g] = st[(b * 8 + g) * 2];
        ig[g] = st[(b * 8 + g) * 2 + 1];
    }

    float accx = 0.0f, accy = 0.0f, accc = 0.0f;
    const float* Yb = Y + (size_t)b * HW * HID;
    for (int t = 0; t < 9; ++t) {
        int yy = y + t / 3 - 1, xx = x + t % 3 - 1;
        if (yy < 0 || yy >= HH || xx < 0 || xx >= WW) continue;
        const float* Yr = Yb + (size_t)(yy * WW + xx) * HID;
        for (int c = 0; c < HID; ++c) {
            float h = gelu_exact((Yr[c] - mg[c >> 3]) * ig[c >> 3] * g3[c] + b3[c]);
            accx += h * dw[c * 9 + t];               // delta ch 0
            accy += h * dw[HID * 9 + c * 9 + t];     // delta ch 1
            accc += h * cfw[c * 9 + t];              // conf  ch 0
        }
    }

    float drx = tanhf(accx + db[0]);
    float dry = tanhf(accy + db[1]);
    float rx = route_clip(cw[2 * p + 0]);
    float ry = route_clip(cw[2 * p + 1]);
    const float scale = 2.0f * 4.0f / 128.0f;        // 0.0625
    float fx = fminf(fmaxf(rx + drx * scale, -1.5f), 1.5f);
    float fy = fminf(fmaxf(ry + dry * scale, -1.5f), 1.5f);
    out[2 * p + 0] = fx;
    out[2 * p + 1] = fy;

    float conf = conf_clip(cf[p]);
    float pc = fminf(fmaxf(conf, 1e-4f), 1.0f - 1e-4f);
    float bl = logf(pc) - log1pf(-pc);
    float logit = bl + 0.5f * (accc + cfb[0]);
    float rc = 1.0f / (1.0f + expf(-logit));
    out[2 * NPIX + p]            = fminf(fmaxf(rc, 0.0f), 1.0f);  // refined_conf
    out[2 * NPIX + NPIX + p]     = logit;                          // refined_logits
}

// ---------------- host ----------------
extern "C" void kernel_launch(void* const* d_in, const int* in_sizes, int n_in,
                              void* d_out, int out_size, void* d_ws, size_t ws_size,
                              hipStream_t stream) {
    const float* fA  = (const float*)d_in[0];
    const float* fB  = (const float*)d_in[1];
    const float* cw  = (const float*)d_in[2];
    const float* cf  = (const float*)d_in[3];
    const float* w1  = (const float*)d_in[4];
    const float* g1  = (const float*)d_in[5];
    const float* b1  = (const float*)d_in[6];
    const float* w2  = (const float*)d_in[7];
    const float* g2  = (const float*)d_in[8];
    const float* b2  = (const float*)d_in[9];
    const float* w3  = (const float*)d_in[10];
    const float* g3  = (const float*)d_in[11];
    const float* b3  = (const float*)d_in[12];
    const float* dw  = (const float*)d_in[13];
    const float* db  = (const float*)d_in[14];
    const float* cfw = (const float*)d_in[15];
    const float* cfb = (const float*)d_in[16];

    char* ws = (char*)d_ws;
    // X buffer (reused as X1 [N][800], then X2/X3 [N][576]) : 32768*800*2 = 52,428,800 B
    _Float16* Xb = (_Float16*)(ws);
    float* Y0 = (float*)(ws + 52428800);             // 32768*64*4 = 8,388,608 B
    float* Y1 = (float*)(ws + 52428800 + 8388608);   // 8,388,608 B
    _Float16* W1 = (_Float16*)(ws + 69206016);       // 64*800*2 = 102,400 B
    _Float16* W2 = (_Float16*)(ws + 69308416);       // 64*576*2 = 73,728 B
    _Float16* W3 = (_Float16*)(ws + 69382144);       // 73,728 B
    float* st1 = (float*)(ws + 69455872);            // 16*(mean,invstd)
    float* st2 = st1 + 32;
    float* st3 = st1 + 64;

    const int prep_total = HID * K1P + HID * K2P;    // one thread fills W2 and W3
    k_prep_weights<<<(prep_total + 255) / 256, 256, 0, stream>>>(w1, w2, w3, W1, W2, W3);

    k_features<<<NPIX / 256, 256, 0, stream>>>(fA, fB, cw, cf, Xb);

    k_gemm<<<NPIX / 32, 256, 0, stream>>>(Xb, W1, Y0, K1P);
    k_stats<<<16, 256, 0, stream>>>(Y0, st1);
    k_act_im2col<<<NPIX / 4, 256, 0, stream>>>(Y0, st1, g1, b1, Xb);

    k_gemm<<<NPIX / 32, 256, 0, stream>>>(Xb, W2, Y1, K2P);
    k_stats<<<16, 256, 0, stream>>>(Y1, st2);
    k_act_im2col<<<NPIX / 4, 256, 0, stream>>>(Y1, st2, g2, b2, Xb);

    k_gemm<<<NPIX / 32, 256, 0, stream>>>(Xb, W3, Y0, K2P);
    k_stats<<<16, 256, 0, stream>>>(Y0, st3);

    k_final<<<NPIX / 256, 256, 0, stream>>>(Y0, st3, g3, b3, cw, cf, dw, db, cfw, cfb,
                                            (float*)d_out);
}
